// SelfAttention_29798483099879
// MI455X (gfx1250) — matrix operand order repaired and checked by reading, not compiled, not run on hardware
//
#include <hip/hip_runtime.h>
#include <cstdint>
#include <cstddef>

// ---------------------------------------------------------------------------
// Self-attention for MI455X (gfx1250, wave32, WMMA + TDM).
//   x:[8,2048,512] f32, W*:[512,512] f32 (torch Linear: y = x@W^T + b)
//   out = softmax((q k^T)/sqrt(H)) v     -> [8,2048,512] f32
//
//   K0: convert x and Wq/Wk/Wv to bf16
//   K1: QKV projection GEMM (v_wmma_f32_16x16x32_bf16), Q pre-scaled by
//       1/sqrt(H), V stored transposed (Vt[b][h][m])
//   K2: flash attention; the block-shared V tile is staged into LDS by the
//       Tensor Data Mover (tensor_load_to_lds, TENSORcnt), double-buffered,
//       so one DMA feeds all 4 waves while WMMAs run on the current tile.
// ---------------------------------------------------------------------------

#define H_DIM   512
#define B_DIM   8
#define N_DIM   2048
#define BN      (B_DIM * N_DIM)
#define RSCALE  0.04419417382415922f   // 1/sqrt(512)

typedef __bf16  bf16_t;
typedef bf16_t  v16bf __attribute__((ext_vector_type(16)));
typedef bf16_t  v8bf  __attribute__((ext_vector_type(8)));
typedef float   v8f   __attribute__((ext_vector_type(8)));
typedef uint32_t u32x4 __attribute__((ext_vector_type(4)));
typedef uint32_t u32x8 __attribute__((ext_vector_type(8)));

__device__ __forceinline__ uint16_t f32_to_bf16(float f) {
    uint32_t u = __float_as_uint(f);
    uint32_t r = u + 0x7FFFu + ((u >> 16) & 1u);   // round to nearest even
    return (uint16_t)(r >> 16);
}

// A-fragment 16x32 bf16 (ISA 7.12.2): lane m=lane&15, half=lane>>4.
// element e -> k = (e/8)*16 + half*8 + (e%8): two contiguous 8-elem chunks.
__device__ __forceinline__ v16bf load_frag_a(const uint16_t* src, int ld) {
    const int lane = threadIdx.x & 31;
    const int m = lane & 15, half = lane >> 4;
    const uint16_t* p = src + (size_t)m * ld + half * 8;
    v8bf c0 = *reinterpret_cast<const v8bf*>(p);
    v8bf c1 = *reinterpret_cast<const v8bf*>(p + 16);
    v16bf a;
#pragma unroll
    for (int i = 0; i < 8; ++i) { a[i] = c0[i]; a[i + 8] = c1[i]; }
    return a;
}

// B-fragment 32x16 bf16: lane holds column n=lane&15, K elems (e/8)*16+half*8+(e%8),
// contiguous when a column of B is a contiguous row of `src` (leading dim ld).
__device__ __forceinline__ v16bf load_frag_b(const uint16_t* src, int ld) {
    const int lane = threadIdx.x & 31;
    const int n = lane & 15, half = lane >> 4;
    const uint16_t* p = src + (size_t)n * ld + half * 8;
    v8bf c0 = *reinterpret_cast<const v8bf*>(p);
    v8bf c1 = *reinterpret_cast<const v8bf*>(p + 16);
    v16bf b;
#pragma unroll
    for (int i = 0; i < 8; ++i) { b[i] = c0[i]; b[i + 8] = c1[i]; }
    return b;
}

__device__ __forceinline__ v8f wmma_bf16(v16bf a, v16bf b, v8f c) {
    return __builtin_amdgcn_wmma_f32_16x16x32_bf16(
        false, a, false, b, (short)0, c, false, false);
}

// ---------------------------------------------------------------------------
// Tensor Data Mover: 2D tile (tile_d0 contiguous elems x tile_d1 rows) of
// 2-byte data from global (row stride stride0 elems) into contiguous LDS.
// D# built per ISA 8.3/8.4: group0 = {flags, lds_addr, global_addr, type=2},
// group1 = {data_size=2B, tensor dims, tile dims, dim0 stride}.
// ---------------------------------------------------------------------------
__device__ __forceinline__ void tdm_load_2d(uint32_t lds_addr, uint64_t gaddr,
                                            uint32_t tile_d0, uint32_t tile_d1,
                                            uint32_t tensor_d0, uint32_t tensor_d1,
                                            uint32_t stride0) {
    u32x4 g0;
    g0[0] = 1u;                                                  // count=1 (valid)
    g0[1] = lds_addr;                                            // LDS byte addr
    g0[2] = (uint32_t)gaddr;                                     // gaddr[31:0]
    g0[3] = ((uint32_t)(gaddr >> 32) & 0x01FFFFFFu) | (2u << 30);// gaddr[56:32]|type=2
    u32x8 g1;
    g1[0] = 1u << 16;                                            // data_size = 2B
    g1[1] = (tensor_d0 & 0xFFFFu) << 16;                         // tensor_dim0 lo
    g1[2] = (tensor_d0 >> 16) | ((tensor_d1 & 0xFFFFu) << 16);   // dim0 hi | dim1 lo
    g1[3] = (tensor_d1 >> 16) | (tile_d0 << 16);                 // dim1 hi | tile_dim0
    g1[4] = tile_d1;                                             // tile_dim1 (dim2=0)
    g1[5] = stride0;                                             // dim0_stride lo
    g1[6] = 0u;
    g1[7] = 0u;
    asm volatile("tensor_load_to_lds %0, %1" :: "s"(g0), "s"(g1) : "memory");
}

// ---------------------------------------------------------------------------
// K0: fp32 -> bf16 conversion of Wq|Wk|Wv (packed) and x.
// ---------------------------------------------------------------------------
__global__ __launch_bounds__(256)
void cvt_bf16_kernel(const float* __restrict__ Wq, const float* __restrict__ Wk,
                     const float* __restrict__ Wv, const float* __restrict__ x,
                     uint16_t* __restrict__ w_dst, uint16_t* __restrict__ x_dst) {
    const size_t HH = (size_t)H_DIM * H_DIM;
    size_t i = (size_t)blockIdx.x * blockDim.x + threadIdx.x;
    const size_t nw = 3 * HH, nx = (size_t)BN * H_DIM;
    if (i < nw) {
        float v = (i < HH) ? Wq[i] : (i < 2 * HH) ? Wk[i - HH] : Wv[i - 2 * HH];
        w_dst[i] = f32_to_bf16(v);
    } else if (i < nw + nx) {
        x_dst[i - nw] = f32_to_bf16(x[i - nw]);
    }
}

// ---------------------------------------------------------------------------
// K1: QKV projection.  grid = (H/64, BN/128, 3), block = 256 (8 waves).
// Each wave: 32 rows x 32 cols = 2x2 WMMA tiles, K loop over H=512.
// z==0 -> Q (scaled), z==1 -> K, z==2 -> V stored transposed Vt[b][h][m].
// ---------------------------------------------------------------------------
__global__ __launch_bounds__(256)
void qkv_kernel(const uint16_t* __restrict__ xbf, const uint16_t* __restrict__ wbf,
                const float* __restrict__ bq, const float* __restrict__ bk,
                const float* __restrict__ bv,
                uint16_t* __restrict__ qb, uint16_t* __restrict__ kb,
                uint16_t* __restrict__ vtb) {
    const int z = blockIdx.z;
    const uint16_t* W = wbf + (size_t)z * H_DIM * H_DIM;

    const int tid = threadIdx.x;
    const int wave = tid >> 5;
    const int wave_m = wave & 3, wave_n = wave >> 2;
    const int R = blockIdx.y * 128 + wave_m * 32;
    const int C = blockIdx.x * 64 + wave_n * 32;

    v8f acc[2][2] = {};
#pragma unroll
    for (int kk = 0; kk < 16; ++kk) {
        const int ko = kk * 32;
        v16bf a0 = load_frag_a(xbf + (size_t)R * H_DIM + ko, H_DIM);
        v16bf a1 = load_frag_a(xbf + (size_t)(R + 16) * H_DIM + ko, H_DIM);
        v16bf b0 = load_frag_b(W + (size_t)C * H_DIM + ko, H_DIM);
        v16bf b1 = load_frag_b(W + (size_t)(C + 16) * H_DIM + ko, H_DIM);
        acc[0][0] = wmma_bf16(a0, b0, acc[0][0]);
        acc[0][1] = wmma_bf16(a0, b1, acc[0][1]);
        acc[1][0] = wmma_bf16(a1, b0, acc[1][0]);
        acc[1][1] = wmma_bf16(a1, b1, acc[1][1]);
    }

    const int lane = tid & 31, half = lane >> 4, ln = lane & 15;
    const float* bias = (z == 0) ? bq : (z == 1) ? bk : bv;
#pragma unroll
    for (int ci = 0; ci < 2; ++ci) {
        const int c = C + ci * 16 + ln;
        const float bb = bias[c];
#pragma unroll
        for (int mi = 0; mi < 2; ++mi) {
            const int rowb = R + mi * 16 + half * 8;
#pragma unroll
            for (int r = 0; r < 8; ++r) {
                float v = acc[mi][ci][r] + bb;
                if (z == 0) v *= RSCALE;            // fold 1/sqrt(H) into Q
                const int rr = rowb + r;            // flat row in [0, B*N)
                if (z == 0) {
                    qb[(size_t)rr * H_DIM + c] = f32_to_bf16(v);
                } else if (z == 1) {
                    kb[(size_t)rr * H_DIM + c] = f32_to_bf16(v);
                } else {
                    const int bidx = rr >> 11, m = rr & (N_DIM - 1);
                    vtb[((size_t)bidx * H_DIM + c) * N_DIM + m] = f32_to_bf16(v);
                }
            }
        }
    }
}

// ---------------------------------------------------------------------------
// K2: flash attention. grid = (BN/64, H/256), block = 128 (4 waves).
// Each wave owns 16 query rows and 256 output columns (16 f32 accum tiles).
// The 256x32 Vt tile (shared by all waves) is TDM-staged into LDS, double
// buffered: wave0 issues tile j+1's DMA, s_wait_tensorcnt 1 guarantees tile j
// landed (TDM in-order per wave), barrier publishes it to the block.
// ---------------------------------------------------------------------------
__global__ __launch_bounds__(128)
void attn_kernel(const uint16_t* __restrict__ qb, const uint16_t* __restrict__ kb,
                 const uint16_t* __restrict__ vtb, float* __restrict__ out) {
    const int tid = threadIdx.x;
    const int wave = tid >> 5, lane = tid & 31;
    const int half = lane >> 4, ln = lane & 15;
    const int wave_u = __builtin_amdgcn_readfirstlane(wave);  // provably uniform

    const int row0  = blockIdx.x * 64 + wave * 16;      // global query-row base
    const int batch = row0 >> 11;                        // row0 / N
    const int hbase = blockIdx.y * 256;                  // output column half

    const uint16_t* Qw  = qb  + (size_t)row0 * H_DIM;
    const uint16_t* Kb  = kb  + ((size_t)batch * N_DIM) * H_DIM;
    const uint16_t* Vtb = vtb + (size_t)batch * H_DIM * N_DIM;

    // LDS: double-buffered Vt tile (2 x 256x32 bf16 = 32KB) + P staging (4KB).
    __shared__ alignas(16) uint16_t vls[2][256 * 32];
    __shared__ alignas(16) uint16_t pls[4][16 * 32];
    uint16_t* myp = &pls[wave][0];

    v8f acc[16] = {};
    float mrun[8], lrun[8];
#pragma unroll
    for (int r = 0; r < 8; ++r) { mrun[r] = -3.0e38f; lrun[r] = 0.0f; }

    // Prologue: stage tile 0.
    if (wave_u == 0) {
        const uint16_t* gp = Vtb + (size_t)hbase * N_DIM;   // rows h, cols m=0..31
        tdm_load_2d((uint32_t)(uintptr_t)&vls[0][0], (uint64_t)(uintptr_t)gp,
                    /*tile_d0=*/32, /*tile_d1=*/256,
                    /*tensor_d0=*/N_DIM, /*tensor_d1=*/H_DIM, /*stride0=*/N_DIM);
    }

    for (int j = 0; j < N_DIM / 32; ++j) {
        const int jb = j * 32;
        const int cur = j & 1;

        if (wave_u == 0) {
            if (j + 1 < N_DIM / 32) {
                const uint16_t* gp = Vtb + (size_t)hbase * N_DIM + (jb + 32);
                tdm_load_2d((uint32_t)(uintptr_t)&vls[cur ^ 1][0],
                            (uint64_t)(uintptr_t)gp,
                            32, 256, N_DIM, H_DIM, N_DIM);
                __builtin_amdgcn_s_wait_tensorcnt(1);   // tile j complete
            } else {
                __builtin_amdgcn_s_wait_tensorcnt(0);
            }
        }
        __syncthreads();                                 // tile j visible to block

        if (j + 1 < N_DIM / 32)                          // global_prefetch_b8
            __builtin_prefetch(Kb + (size_t)(jb + 32) * H_DIM, 0, 0);

        // ---- S tile 16x32 (Q carries 1/sqrt(H)) ----
        v8f s0 = {}, s1 = {};
#pragma unroll
        for (int kk = 0; kk < 16; ++kk) {
            const int ko = kk * 32;
            v16bf aq = load_frag_a(Qw + ko, H_DIM);
            v16bf b0 = load_frag_b(Kb + (size_t)jb * H_DIM + ko, H_DIM);
            v16bf b1 = load_frag_b(Kb + (size_t)(jb + 16) * H_DIM + ko, H_DIM);
            s0 = wmma_bf16(aq, b0, s0);
            s1 = wmma_bf16(aq, b1, s1);
        }

        // ---- online softmax over the 32 new columns ----
        float alphav[8];
#pragma unroll
        for (int r = 0; r < 8; ++r) {
            float t = fmaxf(s0[r], s1[r]);                // row = half*8 + r
            t = fmaxf(t, __shfl_xor(t, 1, 32));
            t = fmaxf(t, __shfl_xor(t, 2, 32));
            t = fmaxf(t, __shfl_xor(t, 4, 32));
            t = fmaxf(t, __shfl_xor(t, 8, 32));           // max over 16 cols
            const float mnew  = fmaxf(mrun[r], t);
            const float alpha = __expf(mrun[r] - mnew);
            const float p0 = __expf(s0[r] - mnew);
            const float p1 = __expf(s1[r] - mnew);
            myp[(half * 8 + r) * 32 + ln]      = f32_to_bf16(p0);
            myp[(half * 8 + r) * 32 + 16 + ln] = f32_to_bf16(p1);
            float rs = p0 + p1;
            rs += __shfl_xor(rs, 1, 32);
            rs += __shfl_xor(rs, 2, 32);
            rs += __shfl_xor(rs, 4, 32);
            rs += __shfl_xor(rs, 8, 32);
            lrun[r] = lrun[r] * alpha + rs;
            mrun[r] = mnew;
            alphav[r] = alpha;
        }
#pragma unroll
        for (int t16 = 0; t16 < 16; ++t16)
#pragma unroll
            for (int r = 0; r < 8; ++r) acc[t16][r] *= alphav[r];

        // P is wave-private in LDS; only an in-wave DS wait is needed.
        asm volatile("s_wait_dscnt 0x0" ::: "memory");
        v16bf pA = load_frag_a(myp, 32);

        // ---- O += P * V from the TDM-staged LDS tile (ld = 32) ----
#pragma unroll
        for (int t16 = 0; t16 < 16; ++t16) {
            v16bf bv = load_frag_b(&vls[cur][(t16 * 16) * 32], 32);
            acc[t16] = wmma_bf16(pA, bv, acc[t16]);
        }

        __syncthreads();   // block done with vls[cur] before it is re-staged
    }

    // ---- normalize and store fp32 output ----
    float invl[8];
#pragma unroll
    for (int r = 0; r < 8; ++r) invl[r] = 1.0f / lrun[r];
#pragma unroll
    for (int t16 = 0; t16 < 16; ++t16) {
        const int c = hbase + t16 * 16 + ln;
#pragma unroll
        for (int r = 0; r < 8; ++r) {
            const int rr = row0 + half * 8 + r;
            out[(size_t)rr * H_DIM + c] = acc[t16][r] * invl[r];
        }
    }
}

// ---------------------------------------------------------------------------
extern "C" void kernel_launch(void* const* d_in, const int* in_sizes, int n_in,
                              void* d_out, int out_size, void* d_ws, size_t ws_size,
                              hipStream_t stream) {
    const float* x  = (const float*)d_in[0];
    const float* Wq = (const float*)d_in[1];
    const float* bq = (const float*)d_in[2];
    const float* Wk = (const float*)d_in[3];
    const float* bk = (const float*)d_in[4];
    const float* Wv = (const float*)d_in[5];
    const float* bv = (const float*)d_in[6];
    float* out = (float*)d_out;

    // Workspace layout (bytes): Wbf (3*H*H*2) | Xbf | Q | K | Vt  (~65.5 MB)
    char* ws = (char*)d_ws;
    const size_t HH  = (size_t)H_DIM * H_DIM;
    const size_t BNH = (size_t)BN * H_DIM;
    uint16_t* wbf = (uint16_t*)(ws);
    uint16_t* xbf = (uint16_t*)(ws + 3 * HH * 2);
    uint16_t* qb  = (uint16_t*)(ws + 3 * HH * 2 + BNH * 2);
    uint16_t* kb  = (uint16_t*)(ws + 3 * HH * 2 + 2 * BNH * 2);
    uint16_t* vtb = (uint16_t*)(ws + 3 * HH * 2 + 3 * BNH * 2);

    {   // K0: convert weights + x to bf16
        const size_t total = 3 * HH + BNH;
        const int blk = 256;
        const int grid = (int)((total + blk - 1) / blk);
        cvt_bf16_kernel<<<grid, blk, 0, stream>>>(Wq, Wk, Wv, x, wbf, xbf);
    }
    {   // K1: QKV projection (Q scaled, V transposed)
        dim3 grid(H_DIM / 64, BN / 128, 3);
        qkv_kernel<<<grid, 256, 0, stream>>>(xbf, wbf, bq, bk, bv, qb, kb, vtb);
    }
    {   // K2: flash attention with TDM-staged V tiles
        dim3 grid(BN / 64, H_DIM / 256);
        attn_kernel<<<grid, 128, 0, stream>>>(qb, kb, vtb, out);
    }
}
